// EarthAttentionBase_16595753631913
// MI455X (gfx1250) — matrix operand order repaired and checked by reading, not compiled, not run on hardware
//
#include <hip/hip_runtime.h>
#include <hip/hip_bf16.h>

// ---------------------------------------------------------------------------
// EarthAttention (Swin-style windowed MHA) forward for gfx1250 / MI455X.
// GEMMs on v_wmma_f32_16x16x32_f16 (wave32 WMMA); kernel-3 A-tile staging uses
// the Tensor Data Mover (tensor_load_to_lds + s_wait_tensorcnt).
// ---------------------------------------------------------------------------

typedef __attribute__((ext_vector_type(16))) _Float16 v16h;
typedef __attribute__((ext_vector_type(8)))  _Float16 v8h;
typedef __attribute__((ext_vector_type(4)))  _Float16 v4h;
typedef __attribute__((ext_vector_type(8)))  float    v8f;
typedef __attribute__((ext_vector_type(4)))  unsigned int v4u;
typedef __attribute__((ext_vector_type(8)))  int      v8i;
typedef __attribute__((ext_vector_type(4)))  int      v4i;
typedef _Float16 h16;

#define BW_    2048
#define NP_    144     // patches per window
#define DIM_   192
#define HEADS_ 6
#define HD_    32      // head dim == WMMA K
#define NWIN_  512
#define TQKV_  576     // 3*DIM

#if __has_builtin(__builtin_amdgcn_tensor_load_to_lds) && \
    __has_builtin(__builtin_amdgcn_s_wait_tensorcnt)
#define USE_TDM 1
#endif

static __device__ __forceinline__ v8f wmma16(v16h a, v16h b, v8f c) {
  // D = A(16x32 f16) * B(32x16 f16) + C(16x16 f32)
  return __builtin_amdgcn_wmma_f32_16x16x32_f16(false, a, false, b,
                                                (short)0, c, false, false);
}

static __device__ __forceinline__ v16h join8(v8h lo, v8h hi) {
  v16h r;
#pragma unroll
  for (int e = 0; e < 8; ++e) { r[e] = lo[e]; r[8 + e] = hi[e]; }
  return r;
}

// ---------------------------------------------------------------------------
// Kernel 0: one-shot f32 -> f16 conversion of w1 / w2.
// ---------------------------------------------------------------------------
__global__ __launch_bounds__(256)
void k_cvt_weights(const float* __restrict__ W1, const float* __restrict__ W2,
                   h16* __restrict__ W1h, h16* __restrict__ W2h) {
  int i = blockIdx.x * 256 + threadIdx.x;
  if (i < TQKV_ * DIM_) W1h[i] = (h16)W1[i];
  if (i < DIM_ * DIM_)  W2h[i] = (h16)W2[i];
}

// ---------------------------------------------------------------------------
// Kernel 1: qkv = x @ w1^T + b1  (store f16, fold softmax scale into q)
// Grid: (BW*NP)/16 blocks of 288 threads (9 waves, 64 output cols each).
// Whole 16x192 X strip staged to LDS once (f32->f16), then a barrier-free,
// fully unrolled K loop of 24 WMMAs.
// ---------------------------------------------------------------------------
__global__ __launch_bounds__(288)
void k_qkv_gemm(const float* __restrict__ X, const h16* __restrict__ W1h,
                const float* __restrict__ B1, h16* __restrict__ QKV) {
  __shared__ h16 As[16][DIM_];                     // 6 KB
  const int wave = threadIdx.x >> 5;
  const int lane = threadIdx.x & 31;
  const int nIdx = lane & 15;
  const int h2   = lane >> 4;
  const int row0 = blockIdx.x * 16;
  const int col0 = wave * 64;
  const float scale = 0.17677669529663687f;        // 32^-0.5

  // stage: 16*192 floats = 768 float4 loads over 288 threads (3 rounds)
#pragma unroll
  for (int it = 0; it < 3; ++it) {
    int i = threadIdx.x + it * 288;
    if (i < 768) {
      int r = i / 48, c4 = (i % 48) * 4;
      float4 xv = *(const float4*)&X[(size_t)(row0 + r) * DIM_ + c4];
      v4h hv; hv[0] = (h16)xv.x; hv[1] = (h16)xv.y;
              hv[2] = (h16)xv.z; hv[3] = (h16)xv.w;
      *(v4h*)&As[r][c4] = hv;
    }
  }
  __syncthreads();

  v8f acc[4];
  v8f zero = {};
#pragma unroll
  for (int t = 0; t < 4; ++t) acc[t] = zero;

#pragma unroll
  for (int kc = 0; kc < DIM_; kc += 32) {
    // A fragment: lane m=nIdx; VGPR0-3: K=8*h2+0..7, VGPR4-7: K=16+8*h2+0..7
    v16h a = join8(*(const v8h*)&As[nIdx][kc + 8 * h2],
                   *(const v8h*)&As[nIdx][kc + 16 + 8 * h2]);
#pragma unroll
    for (int t = 0; t < 4; ++t) {
      int wrow = col0 + 16 * t + nIdx;             // output column
      const h16* wp = W1h + (size_t)wrow * DIM_ + kc + 8 * h2;
      v16h b = join8(*(const v8h*)wp, *(const v8h*)(wp + 16));
      acc[t] = wmma16(a, b, acc[t]);
    }
  }
#pragma unroll
  for (int t = 0; t < 4; ++t) {
    int col = col0 + 16 * t + nIdx;
    float bias = B1[col];
    float s = (col < DIM_) ? scale : 1.0f;         // scale q only
#pragma unroll
    for (int r = 0; r < 8; ++r) {
      int row = row0 + r + 8 * h2;
      QKV[(size_t)row * TQKV_ + col] = (h16)((acc[t][r] + bias) * s);
    }
  }
}

// ---------------------------------------------------------------------------
// Kernel 2: per (bw, head): softmax(q k^T + mask) @ v
// Grid: BW*HEADS blocks of 288 threads; wave i owns score rows 16i..16i+15.
// ---------------------------------------------------------------------------
__global__ __launch_bounds__(288)
void k_attention(const h16* __restrict__ QKV, const float* __restrict__ Mask,
                 h16* __restrict__ OutAtt) {
  __shared__ h16 Vt[32][168];                      // V^T, K padded to 160
  __shared__ h16 Plds[9][16][168];                 // per-wave P staging
  const int bw   = blockIdx.x / HEADS_;
  const int h    = blockIdx.x % HEADS_;
  const int win  = bw & (NWIN_ - 1);
  const int wave = threadIdx.x >> 5;
  const int lane = threadIdx.x & 31;
  const int nIdx = lane & 15;
  const int h2   = lane >> 4;
  const size_t rowbase = (size_t)bw * NP_;

  const float* mbase = Mask + (size_t)win * NP_ * NP_;
  __builtin_prefetch(mbase + (size_t)(wave * 16 + nIdx) * NP_, 0, 3);

  // zero-pad V rows 144..159 (512 elems, counted)
#pragma unroll
  for (int it = 0; it < 2; ++it) {
    int i = threadIdx.x + it * 288;
    if (i < 512) Vt[i >> 4][144 + (i & 15)] = (h16)0.f;
  }
  // stage V^T: 144*32 halfs = 2304 dwords over 288 threads (exactly 8 each)
#pragma unroll
  for (int it = 0; it < 8; ++it) {
    int i = threadIdx.x + it * 288;
    int m = i >> 4, d2 = (i & 15) * 2;
    const h16* vp = QKV + (rowbase + m) * TQKV_ + 2 * DIM_ + h * HD_ + d2;
    h16 v0 = vp[0], v1 = vp[1];
    Vt[d2][m] = v0;
    Vt[d2 + 1][m] = v1;
  }
  __syncthreads();

  const int m0 = wave * 16;
  // q fragment (A layout), loaded straight from global
  const h16* qp = QKV + (rowbase + m0 + nIdx) * TQKV_ + h * HD_;
  v16h qf = join8(*(const v8h*)(qp + 8 * h2),
                  *(const v8h*)(qp + 16 + 8 * h2));

  v8f acc[9];
  v8f zero = {};
#pragma unroll
  for (int t = 0; t < 9; ++t) {
    // B fragment of k^T: lane holds k row (16t+nIdx), dims 8*h2..8*h2+7, 16+8*h2..16+8*h2+7
    const h16* kp = QKV + (rowbase + 16 * t + nIdx) * TQKV_ + DIM_ + h * HD_ + 8 * h2;
    v16h kf = join8(*(const v8h*)kp, *(const v8h*)(kp + 16));
    acc[t] = wmma16(qf, kf, zero);
  }

  // additive window mask
#pragma unroll
  for (int t = 0; t < 9; ++t)
#pragma unroll
    for (int r = 0; r < 8; ++r)
      acc[t][r] += mbase[(size_t)(m0 + r + 8 * h2) * NP_ + 16 * t + nIdx];

  // row softmax: lanes 0-15 / 16-31 hold different rows; xor masks <=8 stay
  // inside each 16-lane group, so the reduction is per-row.
  float rinv[8];
#pragma unroll
  for (int r = 0; r < 8; ++r) {
    float m = acc[0][r];
#pragma unroll
    for (int t = 1; t < 9; ++t) m = fmaxf(m, acc[t][r]);
#pragma unroll
    for (int off = 8; off >= 1; off >>= 1) m = fmaxf(m, __shfl_xor(m, off, 32));
    float s = 0.f;
#pragma unroll
    for (int t = 0; t < 9; ++t) { float e = __expf(acc[t][r] - m); acc[t][r] = e; s += e; }
#pragma unroll
    for (int off = 8; off >= 1; off >>= 1) s += __shfl_xor(s, off, 32);
    rinv[r] = 1.0f / s;
  }

  // normalized P -> LDS (C layout scatter), zero-pad cols 144..159
  h16* P = &Plds[wave][0][0];
#pragma unroll
  for (int r = 0; r < 8; ++r) {
    int prow = (r + 8 * h2) * 168;
#pragma unroll
    for (int t = 0; t < 9; ++t)
      P[prow + 16 * t + nIdx] = (h16)(acc[t][r] * rinv[r]);
    P[prow + 144 + nIdx] = (h16)0.f;
  }
  __syncthreads();

  // out(16x32) = P(16x160) @ V(160x32), K padded with zeros
#pragma unroll
  for (int nt = 0; nt < 2; ++nt) {
    v8f o = zero;
#pragma unroll
    for (int kb = 0; kb < 5; ++kb) {
      const h16* pp = &Plds[wave][nIdx][32 * kb + 8 * h2];
      v16h pf = join8(*(const v8h*)pp, *(const v8h*)(pp + 16));
      const h16* vp = &Vt[16 * nt + nIdx][32 * kb + 8 * h2];
      v16h vf = join8(*(const v8h*)vp, *(const v8h*)(vp + 16));
      o = wmma16(pf, vf, o);
    }
#pragma unroll
    for (int r = 0; r < 8; ++r) {
      int row = m0 + r + 8 * h2;
      OutAtt[(rowbase + row) * DIM_ + h * HD_ + 16 * nt + nIdx] = (h16)o[r];
    }
  }
}

// ---------------------------------------------------------------------------
// Kernel 3: out = attn_out @ w2^T + b2   (f32 output)
// Grid: (BW*NP)/16 blocks of 96 threads (3 waves x 64 cols).
// The whole 16x192 f16 A strip is staged by ONE Tensor Data Mover descriptor
// (tile 192x16, stride 192, 2-byte elements) when available.
// ---------------------------------------------------------------------------
__global__ __launch_bounds__(96)
void k_proj_gemm(const h16* __restrict__ A, const h16* __restrict__ W2h,
                 const float* __restrict__ B2, float* __restrict__ Y) {
  __shared__ h16 As[16][DIM_];                     // 6 KB, dense rows for TDM
  const int wave = threadIdx.x >> 5;
  const int lane = threadIdx.x & 31;
  const int nIdx = lane & 15;
  const int h2   = lane >> 4;
  const int row0 = blockIdx.x * 16;
  const int col0 = wave * 64;

#ifdef USE_TDM
  if (wave == 0) {
    // D# per ISA ch8: 2D tile 192x16 (dim0 x dim1) of 2-byte elements,
    // row stride 192 elements; groups 2/3 zero (dims beyond 2D unused).
    unsigned long long ga = (unsigned long long)(size_t)(A + (size_t)row0 * DIM_);
    unsigned ldsa = (unsigned)(size_t)&As[0][0];   // addr[31:0] = LDS offset
    v4u g0;
    g0[0] = 1u;                                    // count=1, user descriptor
    g0[1] = ldsa;                                  // lds_addr
    g0[2] = (unsigned)ga;                          // global_addr[31:0]
    g0[3] = (unsigned)((ga >> 32) & 0x01FFFFFFu) | (2u << 30); // ga[56:32],type=2
    v8i g1;
    g1[0] = 0x00010000;                            // data_size=1 (2 bytes)
    g1[1] = (int)(192u << 16);                     // tensor_dim0 = 192
    g1[2] = (int)(16u << 16);                      // tensor_dim1 = 16
    g1[3] = (int)(192u << 16);                     // tile_dim0 = 192
    g1[4] = 16;                                    // tile_dim1 = 16
    g1[5] = 192;                                   // tensor_dim0_stride = 192
    g1[6] = 0; g1[7] = 0;
    v4i g2 = {}; v4i g3 = {};
#if defined(__clang_major__) && (__clang_major__ >= 23)
    v8i g4 = {};
    __builtin_amdgcn_tensor_load_to_lds(g0, g1, g2, g3, g4, 0);
#else
    __builtin_amdgcn_tensor_load_to_lds(g0, g1, g2, g3, 0);
#endif
    __builtin_amdgcn_s_wait_tensorcnt(0);
  }
  __syncthreads();
#else
  // fallback: 16*192 halfs = 768 v4h copies over 96 threads (exactly 8 each)
#pragma unroll
  for (int it = 0; it < 8; ++it) {
    int i = threadIdx.x + it * 96;
    int r = i / 48, c4 = (i % 48) * 4;
    *(v4h*)&As[r][c4] = *(const v4h*)&A[(size_t)(row0 + r) * DIM_ + c4];
  }
  __syncthreads();
#endif

  v8f acc[4];
  v8f zero = {};
#pragma unroll
  for (int t = 0; t < 4; ++t) acc[t] = zero;

#pragma unroll
  for (int kc = 0; kc < DIM_; kc += 32) {
    v16h a = join8(*(const v8h*)&As[nIdx][kc + 8 * h2],
                   *(const v8h*)&As[nIdx][kc + 16 + 8 * h2]);
#pragma unroll
    for (int t = 0; t < 4; ++t) {
      int wrow = col0 + 16 * t + nIdx;
      const h16* wp = W2h + (size_t)wrow * DIM_ + kc + 8 * h2;
      v16h b = join8(*(const v8h*)wp, *(const v8h*)(wp + 16));
      acc[t] = wmma16(a, b, acc[t]);
    }
  }
#pragma unroll
  for (int t = 0; t < 4; ++t) {
    int col = col0 + 16 * t + nIdx;
    float bias = B2[col];
#pragma unroll
    for (int r = 0; r < 8; ++r) {
      int row = row0 + r + 8 * h2;
      Y[(size_t)row * DIM_ + col] = acc[t][r] + bias;
    }
  }
}

// ---------------------------------------------------------------------------
extern "C" void kernel_launch(void* const* d_in, const int* in_sizes, int n_in,
                              void* d_out, int out_size, void* d_ws, size_t ws_size,
                              hipStream_t stream) {
  const float* x    = (const float*)d_in[0];
  const float* mask = (const float*)d_in[1];
  const float* w1   = (const float*)d_in[2];
  const float* b1   = (const float*)d_in[3];
  const float* w2   = (const float*)d_in[4];
  const float* b2   = (const float*)d_in[5];
  float* out = (float*)d_out;

  char* ws = (char*)d_ws;
  h16* qkv = (h16*)ws;                                           // 294912 x 576
  ws += (size_t)BW_ * NP_ * TQKV_ * sizeof(h16);
  h16* att = (h16*)ws;                                           // 294912 x 192
  ws += (size_t)BW_ * NP_ * DIM_ * sizeof(h16);
  h16* w1h = (h16*)ws;                                           // 576 x 192
  ws += (size_t)TQKV_ * DIM_ * sizeof(h16);
  h16* w2h = (h16*)ws;                                           // 192 x 192

  k_cvt_weights<<<(TQKV_ * DIM_ + 255) / 256, 256, 0, stream>>>(w1, w2, w1h, w2h);
  k_qkv_gemm <<<(BW_ * NP_) / 16, 288, 0, stream>>>(x, w1h, b1, qkv);
  k_attention<<<BW_ * HEADS_,    288, 0, stream>>>(qkv, mask, att);
  k_proj_gemm<<<(BW_ * NP_) / 16, 96, 0, stream>>>(att, w2h, b2, out);
}
